// Attention_47459388621296
// MI455X (gfx1250) — compile-verified, matrix operand order checked by reading, not run on hardware
//
#include <hip/hip_runtime.h>

typedef __attribute__((ext_vector_type(16))) __bf16 v16bf;
typedef __attribute__((ext_vector_type(8)))  float  v8f;
typedef int v4i_vs __attribute__((vector_size(16)));   // matches builtin param type

static constexpr int kS      = 2048;
static constexpr int kHidden = 5120;
static constexpr int kHeads  = 16;
static constexpr int kQLora  = 1536;
static constexpr int kKvLora = 512;
static constexpr int kNope   = 128;
static constexpr int kRope   = 64;
static constexpr int kQK     = kNope + kRope;   // 192
static constexpr int kV      = 128;

#if defined(__has_builtin)
#if __has_builtin(__builtin_amdgcn_global_load_async_to_lds_b128)
#define USE_ASYNC_LDS 1
#endif
#endif

struct bf16x16_bits { uint4 a, b; };

// Load one 16-lane-row WMMA fragment (A or B operand, 16-bit element layout per
// CDNA5 ISA 7.12.2) from a row-major bf16 matrix. 'row' = matrix row for this
// lane (caller passes tileRow + lane%16). Lanes 0-15 take K 0-7 and 16-23,
// lanes 16-31 take K 8-15 and 24-31 -> two 16-byte loads per lane.
__device__ inline v16bf load_frag(const __bf16* base, int row, int ld) {
    const int half = (threadIdx.x >> 4) & 1;
    const __bf16* p = base + (size_t)row * ld + half * 8;
    bf16x16_bits t;
    t.a = *(const uint4*)(p);
    t.b = *(const uint4*)(p + 16);
    return __builtin_bit_cast(v16bf, t);
}

__device__ inline v8f wmma_bf16(v16bf a, v16bf b, v8f c) {
    return __builtin_amdgcn_wmma_f32_16x16x32_bf16(false, a, false, b,
                                                   (short)0, c, false, false);
}

// 16-byte async copy global->LDS (CDNA5 GLOBAL_LOAD_ASYNC_TO_LDS_B128),
// tracked by ASYNCcnt; falls back to a synchronous VGPR round trip.
__device__ inline void copy_b128_to_lds(const __bf16* g, __bf16* l) {
#ifdef USE_ASYNC_LDS
    __builtin_amdgcn_global_load_async_to_lds_b128((v4i_vs*)g, (v4i_vs*)l, 0, 0);
#else
    *(uint4*)l = *(const uint4*)g;
#endif
}

__device__ inline void wait_async_lds() {
#ifdef USE_ASYNC_LDS
#if __has_builtin(__builtin_amdgcn_s_wait_asynccnt)
    __builtin_amdgcn_s_wait_asynccnt(0);
#else
    asm volatile("s_wait_asynccnt 0" ::: "memory");
#endif
#endif
}

// ---------------------------------------------------------------- convert ---
__global__ void f32_to_bf16_kernel(const float* __restrict__ in,
                                   __bf16* __restrict__ out, size_t n) {
    size_t i = (size_t)blockIdx.x * blockDim.x + threadIdx.x;
    size_t stride = (size_t)gridDim.x * blockDim.x;
    for (; i < n; i += stride) out[i] = (__bf16)in[i];
}

// ---------------------------------------------------------------- rmsnorm ---
__global__ __launch_bounds__(256)
void rmsnorm_bf16_kernel(const float* __restrict__ x, int stride_in, int len,
                         const float* __restrict__ w,
                         __bf16* __restrict__ y, int stride_out) {
    const int row = blockIdx.x;
    const float* xr = x + (size_t)row * stride_in;
    float s = 0.f;
    for (int i = threadIdx.x; i < len; i += 256) { float v = xr[i]; s += v * v; }
#pragma unroll
    for (int off = 16; off; off >>= 1) s += __shfl_xor(s, off, 32);
    __shared__ float red[8];
    if ((threadIdx.x & 31) == 0) red[threadIdx.x >> 5] = s;
    __syncthreads();
    if (threadIdx.x == 0) {
        float t = 0.f;
        for (int i = 0; i < 8; ++i) t += red[i];
        red[0] = rsqrtf(t / (float)len + 1e-6f);
    }
    __syncthreads();
    const float inv = red[0];
    __bf16* yr = y + (size_t)row * stride_out;
    for (int i = threadIdx.x; i < len; i += 256)
        yr[i] = (__bf16)(xr[i] * inv * w[i]);
}

// ------------------------------------------------------------------- gemm ---
// C[M,N](f32) = A[M,K](bf16,row) @ B[N,K](bf16,row)^T  (+ residual*scale)
// Workgroup tile 128x64, 8 waves in a 4x2 grid, each wave owns a 32x32 tile
// (2 A-frags x 2 B-frags = 4 WMMAs per 32-deep k-step). LDS tiles are double
// buffered and filled with async global->LDS b128 copies so the copy of tile
// k+1 overlaps the WMMAs of tile k.
__global__ __launch_bounds__(256)
void gemm_bf16_kernel(const __bf16* __restrict__ A, int lda,
                      const __bf16* __restrict__ B, int ldb,
                      float* __restrict__ C, int ldc, int K,
                      const float* __restrict__ residual, float res_scale) {
    __shared__ __align__(16) __bf16 As[2][128 * 32];
    __shared__ __align__(16) __bf16 Bs[2][64 * 32];
    const int tid  = threadIdx.x;
    const int lane = tid & 31;
    const int wid  = tid >> 5;
    const int wm   = wid & 3;     // 0..3 -> 32-row group of the 128-row tile
    const int wn   = wid >> 2;    // 0..1 -> 32-col group of the 64-col tile
    const int l16  = lane & 15;
    const int m0 = blockIdx.y * 128, n0 = blockIdx.x * 64;

    // Staging map: A tile 128x32 (16 elems/thread), B tile 64x32 (8/thread).
    const __bf16* gA = A + (size_t)(m0 + (tid >> 1)) * lda + (tid & 1) * 16;
    const __bf16* gB = B + (size_t)(n0 + (tid >> 2)) * ldb + (tid & 3) * 8;

    v8f acc[2][2];
#pragma unroll
    for (int i = 0; i < 2; ++i)
#pragma unroll
        for (int j = 0; j < 2; ++j) acc[i][j] = (v8f){0,0,0,0,0,0,0,0};

    const int nk = K / 32;
    // Prologue: stage tile 0 into buffer 0.
    copy_b128_to_lds(gA,     &As[0][tid * 16]);
    copy_b128_to_lds(gA + 8, &As[0][tid * 16 + 8]);
    copy_b128_to_lds(gB,     &Bs[0][tid * 8]);
    wait_async_lds();
    __syncthreads();

    for (int kt = 0; kt < nk; ++kt) {
        const int cur = kt & 1;
        if (kt + 1 < nk) {
            const int nxt = cur ^ 1;
            const __bf16* ga = gA + (kt + 1) * 32;
            const __bf16* gb = gB + (kt + 1) * 32;
            copy_b128_to_lds(ga,     &As[nxt][tid * 16]);
            copy_b128_to_lds(ga + 8, &As[nxt][tid * 16 + 8]);
            copy_b128_to_lds(gb,     &Bs[nxt][tid * 8]);
        }
        v16bf a0 = load_frag(As[cur], wm * 32 + l16,      32);
        v16bf a1 = load_frag(As[cur], wm * 32 + 16 + l16, 32);
        v16bf b0 = load_frag(Bs[cur], wn * 32 + l16,      32);
        v16bf b1 = load_frag(Bs[cur], wn * 32 + 16 + l16, 32);
        acc[0][0] = wmma_bf16(a0, b0, acc[0][0]);
        acc[0][1] = wmma_bf16(a0, b1, acc[0][1]);
        acc[1][0] = wmma_bf16(a1, b0, acc[1][0]);
        acc[1][1] = wmma_bf16(a1, b1, acc[1][1]);
        wait_async_lds();   // next tile landed in the other buffer
        __syncthreads();    // everyone done reading 'cur' before it is reused
    }

    const int half = lane >> 4;
#pragma unroll
    for (int ar = 0; ar < 2; ++ar) {
        const int mb = m0 + wm * 32 + ar * 16 + half * 8;
#pragma unroll
        for (int br = 0; br < 2; ++br) {
            const int nc = n0 + wn * 32 + br * 16 + l16;
#pragma unroll
            for (int r = 0; r < 8; ++r) {
                size_t o = (size_t)(mb + r) * ldc + nc;
                float cv = acc[ar][br][r];
                if (residual) cv += residual[o] * res_scale;
                C[o] = cv;
            }
        }
    }
}

// ------------------------------------------------------------- rope + pack --
// q:(s,16*192) kv:(s,16*256) ckv:(s,576). Writes:
//  q_bf,k_bf: (h,s,192) bf16; k_out:(h,s,192) f32; v_out:(h,s,128) f32;
//  vt_bf:(h,128,2048) bf16 (d-major V for the flash B-operand).
__global__ __launch_bounds__(256)
void rope_pack_kernel(const float* __restrict__ q, const float* __restrict__ kv,
                      const float* __restrict__ ckv,
                      __bf16* __restrict__ q_bf, __bf16* __restrict__ k_bf,
                      float* __restrict__ k_out, __bf16* __restrict__ vt_bf,
                      float* __restrict__ v_out) {
    const int s  = blockIdx.x;
    const int t  = threadIdx.x;
    const int h  = t >> 4;
    const int ln = t & 15;
    const float* qrow  = q   + (size_t)s * (kHeads * kQK) + h * kQK;
    const float* kvrow = kv  + (size_t)s * (kHeads * 256) + h * 256;
    const float* kpe   = ckv + (size_t)s * 576 + kKvLora;
    const size_t hs = (size_t)h * kS + s;
    __bf16* qb = q_bf + hs * kQK;
    __bf16* kb = k_bf + hs * kQK;
    float*  ko = k_out + hs * kQK;
    float*  vo = v_out + hs * kV;
    for (int i = ln; i < kNope; i += 16) {
        float qn = qrow[i];
        qb[i] = (__bf16)qn;
        float kn = kvrow[i];
        kb[i] = (__bf16)kn;
        ko[i] = kn;
        float vv = kvrow[kNope + i];
        vo[i] = vv;
        vt_bf[((size_t)h * kV + i) * kS + s] = (__bf16)vv;
    }
    // RoPE with interleave folded in: out[d]    = x[2d]c - x[2d+1]s
    //                                 out[32+d] = x[2d+1]c + x[2d]s
    for (int d = ln; d < 32; d += 16) {
        float inv_freq = __expf(-(float)(2 * d) * (9.2103403719761836f / 64.0f));
        float f = (float)s * inv_freq;
        float c = __cosf(f), sn = __sinf(f);
        float x0 = qrow[kNope + 2 * d], x1 = qrow[kNope + 2 * d + 1];
        qb[kNope + d]      = (__bf16)(x0 * c - x1 * sn);
        qb[kNope + 32 + d] = (__bf16)(x1 * c + x0 * sn);
        float k0v = kpe[2 * d], k1v = kpe[2 * d + 1];
        float ko0 = k0v * c - k1v * sn;
        float ko1 = k1v * c + k0v * sn;
        kb[kNope + d]      = (__bf16)ko0;
        kb[kNope + 32 + d] = (__bf16)ko1;
        ko[kNope + d]      = ko0;
        ko[kNope + 32 + d] = ko1;
    }
}

// ------------------------------------------------------------ flash attn ----
// grid (32,16), block 128 = 4 waves; wave w handles q-block blockIdx.x*4+w
// (16 queries). Online softmax over 32-key tiles, causal.
__global__ __launch_bounds__(128)
void flash_attn_kernel(const __bf16* __restrict__ q_bf,
                       const __bf16* __restrict__ k_bf,
                       const __bf16* __restrict__ vt_bf,
                       __bf16* __restrict__ attn_bf) {
    const int h    = blockIdx.y;
    const int wid  = threadIdx.x >> 5;
    const int lane = threadIdx.x & 31;
    const int half = lane >> 4;
    const int l16  = lane & 15;
    const int s0   = (blockIdx.x * 4 + wid) * 16;
    __shared__ __align__(16) __bf16 Ps[4][16 * 32];
    const __bf16* Q  = q_bf  + (size_t)h * kS * kQK;
    const __bf16* Km = k_bf  + (size_t)h * kS * kQK;
    const __bf16* Vt = vt_bf + (size_t)h * kV * kS;
    v16bf qf[6];
#pragma unroll
    for (int i = 0; i < 6; ++i)
        qf[i] = load_frag(Q + (size_t)s0 * kQK + i * 32, l16, kQK);
    v8f outacc[8];
#pragma unroll
    for (int t = 0; t < 8; ++t) outacc[t] = (v8f){0,0,0,0,0,0,0,0};
    float mrow[8], lrow[8];
#pragma unroll
    for (int r = 0; r < 8; ++r) { mrow[r] = -1e30f; lrow[r] = 0.f; }
    const float scale = 0.07216878364870323f;  // 192^-0.5
    const int nkb = s0 / 32 + 1;               // causal key-block count
    for (int kb = 0; kb < nkb; ++kb) {
        const int k0 = kb * 32;
        v8f sc0 = {0,0,0,0,0,0,0,0};
        v8f sc1 = {0,0,0,0,0,0,0,0};
#pragma unroll
        for (int i = 0; i < 6; ++i) {
            v16bf kf0 = load_frag(Km + (size_t)k0 * kQK + i * 32, l16, kQK);
            v16bf kf1 = load_frag(Km + (size_t)(k0 + 16) * kQK + i * 32, l16, kQK);
            sc0 = wmma_bf16(qf[i], kf0, sc0);
            sc1 = wmma_bf16(qf[i], kf1, sc1);
        }
        float alpha[8], p0[8], p1[8];
#pragma unroll
        for (int r = 0; r < 8; ++r) {
            const int qrow = s0 + half * 8 + r;
            float v0 = sc0[r] * scale + ((k0 + l16)      > qrow ? -1e9f : 0.f);
            float v1 = sc1[r] * scale + ((k0 + 16 + l16) > qrow ? -1e9f : 0.f);
            float mx = fmaxf(v0, v1);
#pragma unroll
            for (int off = 8; off; off >>= 1) mx = fmaxf(mx, __shfl_xor(mx, off, 16));
            const float mnew = fmaxf(mrow[r], mx);
            const float a  = __expf(mrow[r] - mnew);
            const float e0 = __expf(v0 - mnew);
            const float e1 = __expf(v1 - mnew);
            float sum = e0 + e1;
#pragma unroll
            for (int off = 8; off; off >>= 1) sum += __shfl_xor(sum, off, 16);
            lrow[r] = lrow[r] * a + sum;
            mrow[r] = mnew;
            alpha[r] = a; p0[r] = e0; p1[r] = e1;
        }
        // C-layout -> A-layout transpose of P through this wave's LDS buffer.
        __bf16* P = Ps[wid];
#pragma unroll
        for (int r = 0; r < 8; ++r) {
            P[(half * 8 + r) * 32 + l16]      = (__bf16)p0[r];
            P[(half * 8 + r) * 32 + 16 + l16] = (__bf16)p1[r];
        }
        asm volatile("s_wait_dscnt 0" ::: "memory");
        v16bf pf = load_frag(P, l16, 32);
#pragma unroll
        for (int t = 0; t < 8; ++t) {
            v16bf vf = load_frag(Vt + (size_t)(t * 16) * kS + k0, l16, kS);
#pragma unroll
            for (int r = 0; r < 8; ++r) outacc[t][r] *= alpha[r];
            outacc[t] = wmma_bf16(pf, vf, outacc[t]);
        }
    }
#pragma unroll
    for (int t = 0; t < 8; ++t) {
#pragma unroll
        for (int r = 0; r < 8; ++r) {
            const int m = s0 + half * 8 + r;
            const int d = t * 16 + l16;
            attn_bf[(size_t)m * (kHeads * kV) + h * kV + d] =
                (__bf16)(outacc[t][r] / lrow[r]);
        }
    }
}

// ------------------------------------------------------------------ launch --
extern "C" void kernel_launch(void* const* d_in, const int* in_sizes, int n_in,
                              void* d_out, int out_size, void* d_ws, size_t ws_size,
                              hipStream_t stream) {
    (void)in_sizes; (void)n_in; (void)out_size; (void)ws_size;
    const float* hidden    = (const float*)d_in[0];
    // d_in[1]=position_ids (arange), d_in[2]=attention_mask (causal) — derived in-kernel.
    const float* ln_w      = (const float*)d_in[3];
    const float* wq_a      = (const float*)d_in[4];
    const float* q_a_ln_w  = (const float*)d_in[5];
    const float* wq_b      = (const float*)d_in[6];
    const float* wkv_a     = (const float*)d_in[7];
    const float* kv_a_ln_w = (const float*)d_in[8];
    const float* wkv_b     = (const float*)d_in[9];
    const float* wo        = (const float*)d_in[10];

    float* out_hidden = (float*)d_out;
    float* out_k = out_hidden + (size_t)kS * kHidden;
    float* out_v = out_k + (size_t)kHeads * kS * kQK;

    char* ws = (char*)d_ws;
    size_t off = 0;
    auto carve = [&](size_t bytes) -> void* {
        void* p = ws + off;
        off = (off + bytes + 255) & ~(size_t)255;
        return p;
    };
    __bf16* wq_a_bf  = (__bf16*)carve((size_t)kQLora * kHidden * 2);
    __bf16* wq_b_bf  = (__bf16*)carve((size_t)(kHeads * kQK) * kQLora * 2);
    __bf16* wkv_a_bf = (__bf16*)carve((size_t)(kKvLora + kRope) * kHidden * 2);
    __bf16* wkv_b_bf = (__bf16*)carve((size_t)(kHeads * 256) * kKvLora * 2);
    __bf16* wo_bf    = (__bf16*)carve((size_t)kHidden * (kHeads * kV) * 2);
    __bf16* h_bf     = (__bf16*)carve((size_t)kS * kHidden * 2);
    float*  qa_f     = (float*) carve((size_t)kS * kQLora * 4);
    __bf16* qa_bf    = (__bf16*)carve((size_t)kS * kQLora * 2);
    float*  q_f      = (float*) carve((size_t)kS * kHeads * kQK * 4);
    float*  ckv_f    = (float*) carve((size_t)kS * 576 * 4);
    __bf16* ckv_bf   = (__bf16*)carve((size_t)kS * kKvLora * 2);
    float*  kv_f     = (float*) carve((size_t)kS * kHeads * 256 * 4);
    __bf16* q_bf     = (__bf16*)carve((size_t)kHeads * kS * kQK * 2);
    __bf16* k_bf     = (__bf16*)carve((size_t)kHeads * kS * kQK * 2);
    __bf16* vt_bf    = (__bf16*)carve((size_t)kHeads * kV * kS * 2);
    __bf16* attn_bf  = (__bf16*)carve((size_t)kS * (kHeads * kV) * 2);

    auto cvt = [&](const float* in, __bf16* outp, size_t n) {
        unsigned blocks = (unsigned)((n + 2047) / 2048);
        f32_to_bf16_kernel<<<blocks, 256, 0, stream>>>(in, outp, n);
    };
    cvt(wq_a,  wq_a_bf,  (size_t)kQLora * kHidden);
    cvt(wq_b,  wq_b_bf,  (size_t)(kHeads * kQK) * kQLora);
    cvt(wkv_a, wkv_a_bf, (size_t)(kKvLora + kRope) * kHidden);
    cvt(wkv_b, wkv_b_bf, (size_t)(kHeads * 256) * kKvLora);
    cvt(wo,    wo_bf,    (size_t)kHidden * (kHeads * kV));

    // h = rms_norm(hidden, ln_w) -> bf16
    rmsnorm_bf16_kernel<<<kS, 256, 0, stream>>>(hidden, kHidden, kHidden, ln_w,
                                                h_bf, kHidden);
    // q_a = h @ wq_a^T
    gemm_bf16_kernel<<<dim3(kQLora / 64, kS / 128), 256, 0, stream>>>(
        h_bf, kHidden, wq_a_bf, kHidden, qa_f, kQLora, kHidden, nullptr, 0.f);
    // rms_norm(q_a)
    rmsnorm_bf16_kernel<<<kS, 256, 0, stream>>>(qa_f, kQLora, kQLora, q_a_ln_w,
                                                qa_bf, kQLora);
    // q = qa_n @ wq_b^T
    gemm_bf16_kernel<<<dim3((kHeads * kQK) / 64, kS / 128), 256, 0, stream>>>(
        qa_bf, kQLora, wq_b_bf, kQLora, q_f, kHeads * kQK, kQLora, nullptr, 0.f);
    // ckv = h @ wkv_a^T  (N = 576)
    gemm_bf16_kernel<<<dim3(576 / 64, kS / 128), 256, 0, stream>>>(
        h_bf, kHidden, wkv_a_bf, kHidden, ckv_f, 576, kHidden, nullptr, 0.f);
    // rms_norm(c_kv)  (first 512 of each 576-row)
    rmsnorm_bf16_kernel<<<kS, 256, 0, stream>>>(ckv_f, 576, kKvLora, kv_a_ln_w,
                                                ckv_bf, kKvLora);
    // kv = ckv_n @ wkv_b^T
    gemm_bf16_kernel<<<dim3((kHeads * 256) / 64, kS / 128), 256, 0, stream>>>(
        ckv_bf, kKvLora, wkv_b_bf, kKvLora, kv_f, kHeads * 256, kKvLora,
        nullptr, 0.f);
    // RoPE + pack heads + emit k/v outputs and transposed V
    rope_pack_kernel<<<kS, 256, 0, stream>>>(q_f, kv_f, ckv_f, q_bf, k_bf,
                                             out_k, vt_bf, out_v);
    // causal flash attention
    flash_attn_kernel<<<dim3(kS / 64, kHeads), 128, 0, stream>>>(q_bf, k_bf,
                                                                 vt_bf, attn_bf);
    // out = attn @ wo^T + hidden * (1/8)
    gemm_bf16_kernel<<<dim3(kHidden / 64, kS / 128), 256, 0, stream>>>(
        attn_bf, kHeads * kV, wo_bf, kHeads * kV, out_hidden, kHidden,
        kHeads * kV, hidden, 0.125f);
}
